// basic_HAN_28037546508932
// MI455X (gfx1250) — matrix operand order repaired and checked by reading, not compiled, not run on hardware
//
#include <hip/hip_runtime.h>

typedef __attribute__((ext_vector_type(16))) _Float16 v16h;
typedef __attribute__((ext_vector_type(8)))  _Float16 v8h;
typedef __attribute__((ext_vector_type(8)))  float    v8f;

constexpr int NN   = 50000;   // nodes
constexpr int EE   = 800000;  // edges per metapath
constexpr int MM   = 3;       // metapaths
constexpr int HHD  = 2;       // heads
constexpr int IND  = 128;     // in_dim
constexpr int HID  = 64;      // hidden
constexpr int HEAD = 32;      // per-head out
constexpr int OUTC = 16;      // classes
constexpr int ATT  = 128;     // atten dim
constexpr int NT   = NN / 16; // 3125 row tiles (exact)
constexpr int MH   = MM * HHD;

// ---------------- async global->LDS staging (CDNA5 ASYNCcnt path) ----------------

__device__ __forceinline__ unsigned lds_off_of(const void* p) {
  // generic pointer to a __shared__ object: low 32 bits = wave-relative LDS byte offset
  return (unsigned)(size_t)p;
}

__device__ __forceinline__ void async_stage(const _Float16* __restrict__ g, const _Float16* lds,
                                            int bytes, int tid, int nthreads) {
  unsigned base  = lds_off_of(lds);
  const char* gc = (const char*)g;
  for (int off = tid * 16; off < bytes; off += nthreads * 16) {
    unsigned    l = base + off;
    const void* p = gc + off;
    asm volatile("global_load_async_to_lds_b128 %0, %1, off" ::"v"(l), "v"(p) : "memory");
  }
}

__device__ __forceinline__ void async_fence_barrier() {
  asm volatile("s_wait_asynccnt 0" ::: "memory");
  __syncthreads();
}

// ---------------- WMMA fragment helpers (v_wmma_f32_16x16x32_f16) ----------------

__device__ __forceinline__ v16h load_a_frag(const _Float16* A, int lda, int row0, int k0, int lane) {
  // A row-major [rows][lda]; 16-bit A 16x32 layout:
  // lanes 0-15: M=lane, K = {k0..k0+7, k0+16..k0+23}
  // lanes 16-31: M=lane-16, K = {k0+8..k0+15, k0+24..k0+31}
  int m  = row0 + (lane & 15);
  int kb = k0 + ((lane & 16) ? 8 : 0);
  const _Float16* p = A + (size_t)m * lda + kb;
  union { v16h v; v8h h[2]; } u;
  u.h[0] = *(const v8h*)(p);
  u.h[1] = *(const v8h*)(p + 16);
  return u.v;
}

__device__ __forceinline__ v16h load_b_frag(const _Float16* Bt, int ldb, int k0, int o0, int lane) {
  // Bt is the transposed weight [O][K] row-major: lane holds column N=lane&15,
  // K = {kb..kb+7, kb+16..kb+23}, kb = k0 + 8*half (same as A).
  int n  = o0 + (lane & 15);
  int kb = k0 + ((lane & 16) ? 8 : 0);
  const _Float16* p = Bt + (size_t)n * ldb + kb;
  union { v16h v; v8h h[2]; } u;
  u.h[0] = *(const v8h*)(p);
  u.h[1] = *(const v8h*)(p + 16);
  return u.v;
}

template <int KDIM>
__device__ __forceinline__ v8f gemm_tile(const _Float16* A, int lda, const _Float16* Bt, int ldb,
                                         int row0, int o0, int lane) {
  v8f acc = {};
#pragma unroll
  for (int k = 0; k < KDIM; k += 32) {
    v16h a = load_a_frag(A, lda, row0, k, lane);
    v16h b = load_b_frag(Bt, ldb, k, o0, lane);
    acc = __builtin_amdgcn_wmma_f32_16x16x32_f16(false, a, false, b, (short)0, acc, false, false);
  }
  return acc;
}

// ---------------- prep: f32->f16 convert and weight transpose ----------------

__global__ void convert_kernel(const float* __restrict__ src, _Float16* __restrict__ dst, long n) {
  for (long i = blockIdx.x * (long)blockDim.x + threadIdx.x; i < n;
       i += (long)gridDim.x * blockDim.x)
    dst[i] = (_Float16)src[i];
}

// dst[b][o][k] = src[b][k][o]   (f32 -> f16 transpose, batched)
__global__ void transpose_kernel(const float* __restrict__ src, _Float16* __restrict__ dst,
                                 int batch, int K, int O) {
  long total = (long)batch * K * O;
  for (long i = blockIdx.x * (long)blockDim.x + threadIdx.x; i < total;
       i += (long)gridDim.x * blockDim.x) {
    int  k   = (int)(i % K);
    long rem = i / K;
    int  o   = (int)(rem % O);
    int  b   = (int)(rem / O);
    dst[i] = (_Float16)src[((size_t)b * K + k) * O + o];
  }
}

// ---------------- encoder: h = lrelu(x @ W_enc + b_enc), f16 out ----------------
// block = (mh, n-tile); async-stage A tile (16x128) + full weight (64x128) to LDS;
// 4 waves cover the 4 output column tiles.

__global__ __launch_bounds__(128) void encoder_kernel(const _Float16* __restrict__ xh,
                                                      const _Float16* __restrict__ WencT,
                                                      const float* __restrict__ b_enc,
                                                      _Float16* __restrict__ h16) {
  __shared__ __attribute__((aligned(32))) _Float16 sA[16 * IND];
  __shared__ __attribute__((aligned(32))) _Float16 sB[HID * IND];
  int mh = blockIdx.x / NT;
  int nt = blockIdx.x % NT;
  int tid = threadIdx.x;
  async_stage(xh + (size_t)nt * 16 * IND, sA, 16 * IND * 2, tid, 128);
  async_stage(WencT + (size_t)mh * HID * IND, sB, HID * IND * 2, tid, 128);
  async_fence_barrier();

  int lane = tid & 31;
  int o0   = (tid >> 5) * 16;
  v8f acc = gemm_tile<IND>(sA, IND, sB, IND, 0, o0, lane);

  int   row0 = nt * 16;
  int   col  = o0 + (lane & 15);
  float bias = b_enc[mh * HID + col];
  _Float16* out = h16 + (size_t)mh * NN * HID;
  int rbase = (lane & 16) ? 8 : 0;
#pragma unroll
  for (int r = 0; r < 8; ++r) {
    float v = acc[r] + bias;
    v = v > 0.f ? v : 0.1f * v;
    out[(size_t)(row0 + rbase + r) * HID + col] = (_Float16)v;
  }
}

// ---------------- conv GEMM: xw = h @ W_conv, f32 out ----------------

__global__ __launch_bounds__(128) void conv_kernel(const _Float16* __restrict__ h16,
                                                   const _Float16* __restrict__ WconvT,
                                                   float* __restrict__ xw) {
  __shared__ __attribute__((aligned(32))) _Float16 sA[16 * HID];
  __shared__ __attribute__((aligned(32))) _Float16 sB[HID * HID];
  int mh = blockIdx.x / NT;
  int nt = blockIdx.x % NT;
  int tid = threadIdx.x;
  async_stage(h16 + ((size_t)mh * NN + nt * 16) * HID, sA, 16 * HID * 2, tid, 128);
  async_stage(WconvT + (size_t)mh * HID * HID, sB, HID * HID * 2, tid, 128);
  async_fence_barrier();

  int lane = tid & 31;
  int o0   = (tid >> 5) * 16;
  v8f acc = gemm_tile<HID>(sA, HID, sB, HID, 0, o0, lane);

  int row0 = nt * 16;
  int col  = o0 + (lane & 15);
  float* out = xw + (size_t)mh * NN * HID;
  int rbase = (lane & 16) ? 8 : 0;
#pragma unroll
  for (int r = 0; r < 8; ++r)
    out[(size_t)(row0 + rbase + r) * HID + col] = acc[r];
}

// ---------------- degrees ----------------

__global__ void deg_count_kernel(const long long* __restrict__ ei, float* __restrict__ deg) {
  long t = blockIdx.x * (long)blockDim.x + threadIdx.x;
  if (t >= (long)MM * EE) return;
  int  m = (int)(t / EE);
  long e = t % EE;
  int dst = (int)ei[((long)m * 2 + 1) * EE + e];
  atomicAdd(&deg[(size_t)m * NN + dst], 1.0f);
}

// in place: deg -> rsqrt(deg+1); invd = 1/(deg+1)
__global__ void deg_inv_kernel(float* __restrict__ deg, float* __restrict__ invd) {
  long total = (long)MM * NN;
  for (long i = blockIdx.x * (long)blockDim.x + threadIdx.x; i < total;
       i += (long)gridDim.x * blockDim.x) {
    float d = deg[i] + 1.0f;
    deg[i]  = rsqrtf(d);
    invd[i] = 1.0f / d;
  }
}

// ---------------- edge scatter: agg[dst] += coef * xw[src] ----------------
// one wave of 32 lanes per edge; each lane moves a float4 (H*HID = 128 floats)

__global__ void scatter_kernel(const long long* __restrict__ ei, const float* __restrict__ invs,
                               const float* __restrict__ xw, float* __restrict__ agg) {
  long t = blockIdx.x * (long)blockDim.x + threadIdx.x;
  long total = (long)MM * EE * 32;
  if (t >= total) return;
  int  j  = (int)(t & 31);
  long me = t >> 5;                 // m*EE + e
  int  m  = (int)(me / EE);
  long e  = me % EE;
  int src = (int)ei[((long)m * 2 + 0) * EE + e];
  int dst = (int)ei[((long)m * 2 + 1) * EE + e];
  float coef = invs[(size_t)m * NN + src] * invs[(size_t)m * NN + dst];
  int hh = j >> 4;                  // head
  int d  = (j & 15) * 4;            // channel base
  size_t mhN = (size_t)(m * HHD + hh) * NN;
  const float4 v = *(const float4*)(xw + (mhN + src) * HID + d);
  float* ap = agg + (mhN + dst) * HID + d;
  atomicAdd(ap + 0, v.x * coef);
  atomicAdd(ap + 1, v.y * coef);
  atomicAdd(ap + 2, v.z * coef);
  atomicAdd(ap + 3, v.w * coef);
}

// ---------------- hh = lrelu(agg + xw/deg + b_conv), f16 out ----------------

__global__ void finalize_kernel(const float* __restrict__ agg, const float* __restrict__ xw,
                                const float* __restrict__ invd, const float* __restrict__ b_conv,
                                _Float16* __restrict__ hh16) {
  long total = (long)MH * NN * HID;
  for (long i = blockIdx.x * (long)blockDim.x + threadIdx.x; i < total;
       i += (long)gridDim.x * blockDim.x) {
    int  d  = (int)(i & 63);
    long nd = i >> 6;               // mh*NN + n
    int  mh = (int)(nd / NN);
    int  n  = (int)(nd % NN);
    int  m  = mh >> 1;
    float v = agg[i] + xw[i] * invd[(size_t)m * NN + n] + b_conv[mh * HID + d];
    v = v > 0.f ? v : 0.1f * v;
    hh16[i] = (_Float16)v;
  }
}

// ---------------- decoder GEMM: zs[m][n][h*32+o] = hh @ W_dec + b_dec ----------------
// all 6 per-(m,h) weights are only 24KB -> each block stages the full set to LDS

__global__ __launch_bounds__(256) void dec_kernel(const _Float16* __restrict__ hh16,
                                                  const _Float16* __restrict__ WdecT,
                                                  const float* __restrict__ b_dec,
                                                  float* __restrict__ zs,
                                                  _Float16* __restrict__ zs16) {
  __shared__ __attribute__((aligned(32))) _Float16 sB[MH * HEAD * HID];
  int tid = threadIdx.x;
  async_stage(WdecT, sB, MH * HEAD * HID * 2, tid, 256);
  async_fence_barrier();

  int wave = (int)((blockIdx.x * (long)blockDim.x + tid) >> 5);
  int lane = tid & 31;
  int total = MH * NT * 2;          // 2 column tiles over HEAD=32
  if (wave >= total) return;
  int ot = wave & 1;
  int nt = (wave >> 1) % NT;
  int mh = wave / (2 * NT);
  int m = mh >> 1, hd = mh & 1;
  const _Float16* A  = hh16 + (size_t)mh * NN * HID;
  const _Float16* Bt = sB + (size_t)mh * HEAD * HID;
  int row0 = nt * 16, o0 = ot * 16;
  v8f acc = gemm_tile<HID>(A, HID, Bt, HID, row0, o0, lane);
  int oc   = o0 + (lane & 15);
  float bias = b_dec[mh * HEAD + oc];
  int col  = hd * HEAD + oc;        // head concat position within 64
  float*    zo   = zs   + (size_t)m * NN * HID;
  _Float16* zo16 = zs16 + (size_t)m * NN * HID;
  int rbase = (lane & 16) ? 8 : 0;
#pragma unroll
  for (int r = 0; r < 8; ++r) {
    float v = acc[r] + bias;
    size_t idx = (size_t)(row0 + rbase + r) * HID + col;
    zo[idx]   = v;
    zo16[idx] = (_Float16)v;
  }
}

// ---------------- semantic attention scores: ssum[m] += sum_n tanh(zs@W_s+b_s)·q ----------------
// W_s is shared by every (m, n-tile) -> stage 16KB once per block

__global__ __launch_bounds__(256) void attn_kernel(const _Float16* __restrict__ zs16,
                                                   const _Float16* __restrict__ WsT,
                                                   const float* __restrict__ b_s,
                                                   const float* __restrict__ q,
                                                   float* __restrict__ ssum) {
  __shared__ __attribute__((aligned(32))) _Float16 sB[ATT * HID];
  int tid = threadIdx.x;
  async_stage(WsT, sB, ATT * HID * 2, tid, 256);
  async_fence_barrier();

  int wave = (int)((blockIdx.x * (long)blockDim.x + tid) >> 5);
  int lane = tid & 31;
  if (wave >= MM * NT) return;
  int nt = wave % NT;
  int m  = wave / NT;
  const _Float16* A = zs16 + (size_t)m * NN * HID;
  int row0 = nt * 16;
  float lsum = 0.f;
#pragma unroll
  for (int at = 0; at < ATT / 16; ++at) {
    int a0 = at * 16;
    v8f acc = gemm_tile<HID>(A, HID, sB, HID, row0, a0, lane);
    int col = a0 + (lane & 15);
    float bias = b_s[col], qa = q[col];
#pragma unroll
    for (int r = 0; r < 8; ++r) lsum += tanhf(acc[r] + bias) * qa;
  }
  for (int off = 16; off; off >>= 1) lsum += __shfl_xor(lsum, off, 32);
  if (lane == 0) atomicAdd(&ssum[m], lsum);
}

// ---------------- beta = softmax(mean score) ----------------

__global__ void beta_kernel(const float* __restrict__ ssum, float* __restrict__ beta_out) {
  if (blockIdx.x == 0 && threadIdx.x == 0) {
    float s0 = ssum[0] / (float)NN, s1 = ssum[1] / (float)NN, s2 = ssum[2] / (float)NN;
    float mx = fmaxf(s0, fmaxf(s1, s2));
    float e0 = expf(s0 - mx), e1 = expf(s1 - mx), e2 = expf(s2 - mx);
    float t = e0 + e1 + e2;
    beta_out[0] = e0 / t; beta_out[1] = e1 / t; beta_out[2] = e2 / t;
  }
}

// ---------------- Z = sum_m beta[m]*zs[m], f16 out ----------------

__global__ void combine_kernel(const float* __restrict__ zs, const float* __restrict__ beta,
                               _Float16* __restrict__ Zh) {
  long total = (long)NN * HID;
  float b0 = beta[0], b1 = beta[1], b2 = beta[2];
  for (long i = blockIdx.x * (long)blockDim.x + threadIdx.x; i < total;
       i += (long)gridDim.x * blockDim.x)
    Zh[i] = (_Float16)(b0 * zs[i] + b1 * zs[total + i] + b2 * zs[2 * total + i]);
}

// ---------------- out = log_softmax(Z @ W_o + b_o) ----------------

__global__ __launch_bounds__(256) void out_kernel(const _Float16* __restrict__ Zh,
                                                  const _Float16* __restrict__ WoT,
                                                  const float* __restrict__ b_o,
                                                  float* __restrict__ out) {
  __shared__ __attribute__((aligned(32))) _Float16 sB[OUTC * HID];
  int tid = threadIdx.x;
  async_stage(WoT, sB, OUTC * HID * 2, tid, 256);
  async_fence_barrier();

  int wave = (int)((blockIdx.x * (long)blockDim.x + tid) >> 5);
  int lane = tid & 31;
  if (wave >= NT) return;
  int row0 = wave * 16;
  v8f acc = gemm_tile<HID>(Zh, HID, sB, HID, row0, 0, lane);
  int col = lane & 15;
  float bias = b_o[col];
  int rbase = (lane & 16) ? 8 : 0;
#pragma unroll
  for (int r = 0; r < 8; ++r) {
    float v = acc[r] + bias;
    // row's 16 class values live in one 16-lane half-wave -> xor-reduce within it
    float mx = v;
    for (int off = 8; off; off >>= 1) mx = fmaxf(mx, __shfl_xor(mx, off, 32));
    float e = expf(v - mx), s = e;
    for (int off = 8; off; off >>= 1) s += __shfl_xor(s, off, 32);
    out[(size_t)(row0 + rbase + r) * OUTC + col] = (v - mx) - logf(s);
  }
}

// ---------------- host launch ----------------

extern "C" void kernel_launch(void* const* d_in, const int* in_sizes, int n_in,
                              void* d_out, int out_size, void* d_ws, size_t ws_size,
                              hipStream_t stream) {
  const float*     x      = (const float*)d_in[0];
  const long long* ei     = (const long long*)d_in[1];
  const float*     W_enc  = (const float*)d_in[2];
  const float*     b_enc  = (const float*)d_in[3];
  const float*     W_conv = (const float*)d_in[4];
  const float*     b_conv = (const float*)d_in[5];
  const float*     W_dec  = (const float*)d_in[6];
  const float*     b_dec  = (const float*)d_in[7];
  const float*     W_s    = (const float*)d_in[8];
  const float*     b_s    = (const float*)d_in[9];
  const float*     q      = (const float*)d_in[10];
  const float*     W_o    = (const float*)d_in[11];
  const float*     b_o    = (const float*)d_in[12];
  float* out = (float*)d_out;                      // [N*16] then beta [3]

  char* w = (char*)d_ws;
  auto alloc = [&](size_t bytes) -> char* {
    char* p = w;
    w += (bytes + 255) & ~(size_t)255;
    return p;
  };

  _Float16* xh     = (_Float16*)alloc((size_t)NN * IND * 2);        // 12.8 MB (reused for Zh)
  _Float16* WencT  = (_Float16*)alloc((size_t)MH * HID * IND * 2);
  _Float16* WconvT = (_Float16*)alloc((size_t)MH * HID * HID * 2);
  _Float16* WdecT  = (_Float16*)alloc((size_t)MH * HEAD * HID * 2);
  _Float16* WsT    = (_Float16*)alloc((size_t)ATT * HID * 2);
  _Float16* WoT    = (_Float16*)alloc((size_t)OUTC * HID * 2);
  _Float16* h16    = (_Float16*)alloc((size_t)MH * NN * HID * 2);   // 38.4 MB (reused for hh16)
  float*    xw     = (float*)alloc((size_t)MH * NN * HID * 4);      // 76.8 MB
  float*    agg    = (float*)alloc((size_t)MH * NN * HID * 4);      // 76.8 MB (reused for zs16)
  float*    invs   = (float*)alloc((size_t)MM * NN * 4);            // deg -> rsqrt in place
  float*    invd   = (float*)alloc((size_t)MM * NN * 4);
  float*    zs     = (float*)alloc((size_t)MM * NN * HID * 4);      // 38.4 MB
  float*    ssum   = (float*)alloc(256);
  _Float16* hh16   = h16;                    // dead after conv GEMM
  _Float16* zs16   = (_Float16*)agg;         // dead after finalize
  _Float16* Zh     = xh;                     // dead after encoder
  float*    beta   = out + (size_t)NN * OUTC;

  const int B = 256;

  // prep: converts + weight transposes
  convert_kernel<<<25000, B, 0, stream>>>(x, xh, (long)NN * IND);
  transpose_kernel<<<384, B, 0, stream>>>(W_enc, WencT, MH, IND, HID);
  transpose_kernel<<<96, B, 0, stream>>>(W_conv, WconvT, MH, HID, HID);
  transpose_kernel<<<48, B, 0, stream>>>(W_dec, WdecT, MH, HID, HEAD);
  transpose_kernel<<<32, B, 0, stream>>>(W_s, WsT, 1, HID, ATT);
  transpose_kernel<<<4, B, 0, stream>>>(W_o, WoT, 1, HID, OUTC);

  // zero accumulators (capture-legal)
  hipMemsetAsync(agg, 0, (size_t)MH * NN * HID * 4, stream);
  hipMemsetAsync(invs, 0, (size_t)MM * NN * 4, stream);
  hipMemsetAsync(ssum, 0, 256, stream);

  // encoder GEMM (WMMA, async-to-LDS staged A+B); block = (mh, n-tile), 4 waves
  encoder_kernel<<<MH * NT, 128, 0, stream>>>(xh, WencT, b_enc, h16);
  // conv GEMM (WMMA, async-to-LDS staged A+B)
  conv_kernel<<<MH * NT, 128, 0, stream>>>(h16, WconvT, xw);
  // degrees
  deg_count_kernel<<<((long)MM * EE + B - 1) / B, B, 0, stream>>>(ei, invs);
  deg_inv_kernel<<<588, B, 0, stream>>>(invs, invd);
  // edge scatter-aggregate
  scatter_kernel<<<(size_t)((long)MM * EE * 32 + B - 1) / B, B, 0, stream>>>(ei, invs, xw, agg);
  // finalize conv + lrelu -> f16
  finalize_kernel<<<75000, B, 0, stream>>>(agg, xw, invd, b_conv, hh16);
  // decoder GEMM (WMMA, staged weights), head concat -> zs (f32) and zs16 (f16)
  dec_kernel<<<(MH * NT * 2 * 32 + B - 1) / B, B, 0, stream>>>(hh16, WdecT, b_dec, zs, zs16);
  // semantic attention scores (WMMA, staged W_s + wave reduce)
  attn_kernel<<<(MM * NT * 32 + B - 1) / B, B, 0, stream>>>(zs16, WsT, b_s, q, ssum);
  // beta softmax -> tail of d_out
  beta_kernel<<<1, 64, 0, stream>>>(ssum, beta);
  // Z = sum_m beta[m]*zs[m]
  combine_kernel<<<12500, B, 0, stream>>>(zs, beta, Zh);
  // output GEMM + log_softmax (WMMA, staged W_o)
  out_kernel<<<(NT * 32 + B - 1) / B, B, 0, stream>>>(Zh, WoT, b_o, out);
}
